// QuantizedLinear_38946763441064
// MI455X (gfx1250) — compile-verified, matrix operand order checked by reading, not run on hardware
//
#include <hip/hip_runtime.h>

typedef __attribute__((ext_vector_type(16))) _Float16 v16h;
typedef __attribute__((ext_vector_type(8)))  _Float16 v8h;
typedef __attribute__((ext_vector_type(8)))  float    v8f;

#define K_DIM 4096
#define N_DIM 4096
#define M_DIM 16384
#define BM 128
#define BN 128

// ---------------------------------------------------------------------------
// Pre-convert kernels: int32 -> f16 (lossless for 0..127) and fp32 -> f16
// ---------------------------------------------------------------------------
__global__ __launch_bounds__(256)
void convert_w_f16(const int* __restrict__ wq, _Float16* __restrict__ w16)
{
    const size_t idx = ((size_t)blockIdx.x * 256 + threadIdx.x) * 8;
    int4 a = *(const int4*)(wq + idx);
    int4 b = *(const int4*)(wq + idx + 4);
    v8h h;
    h[0] = (_Float16)(float)a.x; h[1] = (_Float16)(float)a.y;
    h[2] = (_Float16)(float)a.z; h[3] = (_Float16)(float)a.w;
    h[4] = (_Float16)(float)b.x; h[5] = (_Float16)(float)b.y;
    h[6] = (_Float16)(float)b.z; h[7] = (_Float16)(float)b.w;
    *(v8h*)(w16 + idx) = h;
}

__global__ __launch_bounds__(256)
void convert_x_f16(const float* __restrict__ x, _Float16* __restrict__ x16)
{
    const size_t idx = ((size_t)blockIdx.x * 256 + threadIdx.x) * 8;
    float4 a = *(const float4*)(x + idx);
    float4 b = *(const float4*)(x + idx + 4);
    v8h h;
    h[0] = (_Float16)a.x; h[1] = (_Float16)a.y;
    h[2] = (_Float16)a.z; h[3] = (_Float16)a.w;
    h[4] = (_Float16)b.x; h[5] = (_Float16)b.y;
    h[6] = (_Float16)b.z; h[7] = (_Float16)b.w;
    *(v8h*)(x16 + idx) = h;
}

// ---------------------------------------------------------------------------
// Async global -> LDS helpers (CDNA5 GLOBAL_LOAD_ASYNC_TO_LDS_*, ASYNCcnt)
// INST_OFFSET is added to BOTH the LDS and the global address, so one
// (lds, global) base pair covers a contiguous 64B region with 4 ops.
// ---------------------------------------------------------------------------
__device__ __forceinline__ void async_copy_64B(unsigned lds_off, const void* gptr)
{
    unsigned long long ga = (unsigned long long)(uintptr_t)gptr;
    asm volatile(
        "global_load_async_to_lds_b128 %0, %1, off\n\t"
        "global_load_async_to_lds_b128 %0, %1, off offset:16\n\t"
        "global_load_async_to_lds_b128 %0, %1, off offset:32\n\t"
        "global_load_async_to_lds_b128 %0, %1, off offset:48"
        :: "v"(lds_off), "v"(ga) : "memory");
}

__device__ __forceinline__ void wait_async_0()
{
    asm volatile("s_wait_asynccnt 0x0" ::: "memory");
}

// ---------------------------------------------------------------------------
// Main GEMM on pre-converted f16 operands.
// BK=64, LDS row stride 72 halfs, double-buffered LDS fed by async copies.
// Block 256 threads = 8 waves; block tile 128x128; wave tile 32x64.
// ---------------------------------------------------------------------------
#define GBK  64
#define GLD  72                    // halfs per LDS row (64 + 8 pad)
#define GNIT (K_DIM / GBK)         // 64 K-tiles
#define ABUF (BM * GLD)            // halfs per A buffer (9216 -> 18432 B)
#define BBUF (BN * GLD)

__global__ __launch_bounds__(256)
void gemm_f16_wmma(const _Float16* __restrict__ A,    // x16 [M,K]
                   const _Float16* __restrict__ Bm,   // w16 [N,K]
                   const float*    __restrict__ scale,
                   const float*    __restrict__ bias,
                   float*          __restrict__ out)
{
    __shared__ _Float16 sA[2 * ABUF];   // 36864 B
    __shared__ _Float16 sB[2 * BBUF];   // 36864 B

    const int tid  = threadIdx.x;
    const int lane = tid & 31;
    const int wave = tid >> 5;
    const int wm   = wave & 3;          // M offset = wm*32
    const int wn   = wave >> 2;         // N offset = wn*64

    const int bm0 = (blockIdx.x >> 5) * BM;
    const int bn0 = (blockIdx.x & 31) * BN;

    // copy mapping: tile = 128 rows x 64 halfs; 2 threads per row,
    // each thread owns one contiguous 64B region (global and LDS).
    const int grow = tid >> 1;          // 0..127
    const int hcol = (tid & 1) * 32;    // half offset within 64-wide K slice

    const _Float16* ag = A  + (size_t)(bm0 + grow) * K_DIM + hcol;
    const _Float16* bg = Bm + (size_t)(bn0 + grow) * K_DIM + hcol;

    // LDS byte offsets (low 32 bits of the flat shared-memory address)
    const unsigned ldsA0 = (unsigned)(uintptr_t)&sA[grow * GLD + hcol];
    const unsigned ldsB0 = (unsigned)(uintptr_t)&sB[grow * GLD + hcol];
    const unsigned ldsA1 = ldsA0 + ABUF * sizeof(_Float16);
    const unsigned ldsB1 = ldsB0 + BBUF * sizeof(_Float16);

    v8f acc[2][4];
    #pragma unroll
    for (int i = 0; i < 2; ++i)
        #pragma unroll
        for (int j = 0; j < 4; ++j)
            acc[i][j] = (v8f){0.f, 0.f, 0.f, 0.f, 0.f, 0.f, 0.f, 0.f};

    const int mrow = lane & 15;
    const int kh   = lane >> 4;

    // prologue: async-fill buffer 0 with K-tile 0
    async_copy_64B(ldsA0, ag);
    async_copy_64B(ldsB0, bg);
    wait_async_0();
    __syncthreads();

    for (int it = 0; it < GNIT; ++it) {
        // kick off next tile into the other buffer; overlaps the WMMAs below
        if (it + 1 < GNIT) {
            const size_t k0 = (size_t)(it + 1) * GBK;
            const unsigned la = ((it + 1) & 1) ? ldsA1 : ldsA0;
            const unsigned lb = ((it + 1) & 1) ? ldsB1 : ldsB0;
            async_copy_64B(la, ag + k0);
            async_copy_64B(lb, bg + k0);
        }

        const _Float16* bufA = &sA[(it & 1) * ABUF];
        const _Float16* bufB = &sB[(it & 1) * BBUF];

        // two 32-deep K-chunks per LDS tile
        #pragma unroll
        for (int c = 0; c < 2; ++c) {
            const int kbase = c * 32 + kh * 8;
            v16h af[2];
            #pragma unroll
            for (int i = 0; i < 2; ++i) {
                const _Float16* p = &bufA[(wm * 32 + i * 16 + mrow) * GLD + kbase];
                union { v16h v; v8h h[2]; } u;
                u.h[0] = *(const v8h*)(p);        // K + 0..7
                u.h[1] = *(const v8h*)(p + 16);   // K + 16..23
                af[i] = u.v;
            }
            v16h bf[4];
            #pragma unroll
            for (int j = 0; j < 4; ++j) {
                const _Float16* p = &bufB[(wn * 64 + j * 16 + mrow) * GLD + kbase];
                union { v16h v; v8h h[2]; } u;
                u.h[0] = *(const v8h*)(p);
                u.h[1] = *(const v8h*)(p + 16);
                bf[j] = u.v;
            }
            #pragma unroll
            for (int i = 0; i < 2; ++i)
                #pragma unroll
                for (int j = 0; j < 4; ++j)
                    acc[i][j] = __builtin_amdgcn_wmma_f32_16x16x32_f16(
                        false, af[i], false, bf[j],
                        (short)0, acc[i][j], false, false);
        }

        // this wave's prefetch must have landed; then all waves rendezvous
        if (it + 1 < GNIT)
            wait_async_0();
        __syncthreads();
    }

    const float s = scale[0];
    #pragma unroll
    for (int j = 0; j < 4; ++j) {
        const int gcol = bn0 + wn * 64 + j * 16 + mrow;
        const float bj = bias[gcol];
        #pragma unroll
        for (int i = 0; i < 2; ++i)
            #pragma unroll
            for (int r = 0; r < 8; ++r) {
                const int gm = bm0 + wm * 32 + i * 16 + kh * 8 + r;
                out[(size_t)gm * N_DIM + gcol] = acc[i][j][r] * s + bj;
            }
    }
}

// ---------------------------------------------------------------------------
// Fallback: fully fused kernel (converts in the K-loop), used if the
// workspace is too small for the pre-converted operands.
// ---------------------------------------------------------------------------
#define BK  32
#define LDA 40
#define LDB 40
#define NIT (K_DIM / BK)

__global__ __launch_bounds__(256)
void qlinear_fused(const float* __restrict__ x,
                   const int*   __restrict__ wq,
                   const float* __restrict__ scale,
                   const float* __restrict__ bias,
                   float*       __restrict__ out)
{
    __shared__ _Float16 sA[BM * LDA];
    __shared__ _Float16 sB[BN * LDB];

    const int tid  = threadIdx.x;
    const int lane = tid & 31;
    const int wave = tid >> 5;
    const int wm   = wave & 3;
    const int wn   = wave >> 2;

    const int bm0 = (blockIdx.x >> 5) * BM;
    const int bn0 = (blockIdx.x & 31) * BN;

    const int grow = tid >> 3;
    const int col4 = tid & 7;

    const float* xg = x  + (size_t)(bm0 + grow) * K_DIM + col4 * 4;
    const int*   wg = wq + (size_t)(bn0 + grow) * K_DIM + col4 * 4;

    float4 ra[4];
    int4   rb[4];
    #pragma unroll
    for (int i = 0; i < 4; ++i) {
        ra[i] = *(const float4*)(xg + (size_t)(i * 32) * K_DIM);
        rb[i] = *(const int4*)  (wg + (size_t)(i * 32) * K_DIM);
    }

    v8f acc[2][4];
    #pragma unroll
    for (int i = 0; i < 2; ++i)
        #pragma unroll
        for (int j = 0; j < 4; ++j)
            acc[i][j] = (v8f){0.f, 0.f, 0.f, 0.f, 0.f, 0.f, 0.f, 0.f};

    const int mrow = lane & 15;
    const int kh   = lane >> 4;

    for (int it = 0; it < NIT; ++it) {
        __syncthreads();

        #pragma unroll
        for (int i = 0; i < 4; ++i) {
            const int r = grow + i * 32;
            _Float16* pa = &sA[r * LDA + col4 * 4];
            pa[0] = (_Float16)ra[i].x; pa[1] = (_Float16)ra[i].y;
            pa[2] = (_Float16)ra[i].z; pa[3] = (_Float16)ra[i].w;
            _Float16* pb = &sB[r * LDB + col4 * 4];
            pb[0] = (_Float16)(float)rb[i].x; pb[1] = (_Float16)(float)rb[i].y;
            pb[2] = (_Float16)(float)rb[i].z; pb[3] = (_Float16)(float)rb[i].w;
        }

        __syncthreads();

        if (it + 1 < NIT) {
            const size_t k0 = (size_t)(it + 1) * BK;
            #pragma unroll
            for (int i = 0; i < 4; ++i) {
                ra[i] = *(const float4*)(xg + (size_t)(i * 32) * K_DIM + k0);
                rb[i] = *(const int4*)  (wg + (size_t)(i * 32) * K_DIM + k0);
            }
        }

        v16h af[2];
        #pragma unroll
        for (int i = 0; i < 2; ++i) {
            const _Float16* p = &sA[(wm * 32 + i * 16 + mrow) * LDA + kh * 8];
            union { v16h v; v8h h[2]; } u;
            u.h[0] = *(const v8h*)(p);
            u.h[1] = *(const v8h*)(p + 16);
            af[i] = u.v;
        }
        v16h bf[4];
        #pragma unroll
        for (int j = 0; j < 4; ++j) {
            const _Float16* p = &sB[(wn * 64 + j * 16 + mrow) * LDB + kh * 8];
            union { v16h v; v8h h[2]; } u;
            u.h[0] = *(const v8h*)(p);
            u.h[1] = *(const v8h*)(p + 16);
            bf[j] = u.v;
        }
        #pragma unroll
        for (int i = 0; i < 2; ++i)
            #pragma unroll
            for (int j = 0; j < 4; ++j)
                acc[i][j] = __builtin_amdgcn_wmma_f32_16x16x32_f16(
                    false, af[i], false, bf[j],
                    (short)0, acc[i][j], false, false);
    }

    const float s = scale[0];
    #pragma unroll
    for (int j = 0; j < 4; ++j) {
        const int gcol = bn0 + wn * 64 + j * 16 + mrow;
        const float bj = bias[gcol];
        #pragma unroll
        for (int i = 0; i < 2; ++i)
            #pragma unroll
            for (int r = 0; r < 8; ++r) {
                const int gm = bm0 + wm * 32 + i * 16 + kh * 8 + r;
                out[(size_t)gm * N_DIM + gcol] = acc[i][j][r] * s + bj;
            }
    }
}

// ---------------------------------------------------------------------------
extern "C" void kernel_launch(void* const* d_in, const int* in_sizes, int n_in,
                              void* d_out, int out_size, void* d_ws, size_t ws_size,
                              hipStream_t stream) {
    (void)in_sizes; (void)n_in; (void)out_size;
    const float* x     = (const float*)d_in[0];
    const int*   wq    = (const int*)  d_in[1];
    const float* scale = (const float*)d_in[2];
    const float* bias  = (const float*)d_in[3];
    float*       out   = (float*)d_out;

    const size_t wbytes = (size_t)N_DIM * K_DIM * sizeof(_Float16);   //  32 MB
    const size_t xbytes = (size_t)M_DIM * K_DIM * sizeof(_Float16);   // 128 MB
    const int    grid   = (M_DIM / BM) * (N_DIM / BN);                // 4096

    if (ws_size >= wbytes + xbytes) {
        _Float16* w16 = (_Float16*)d_ws;
        _Float16* x16 = (_Float16*)((char*)d_ws + wbytes);

        const int wblk = (int)(((size_t)N_DIM * K_DIM / 8) / 256);    //  8192
        const int xblk = (int)(((size_t)M_DIM * K_DIM / 8) / 256);    // 32768
        convert_w_f16<<<wblk, 256, 0, stream>>>(wq, w16);
        convert_x_f16<<<xblk, 256, 0, stream>>>(x, x16);

        gemm_f16_wmma<<<grid, 256, 0, stream>>>(x16, w16, scale, bias, out);
    } else {
        qlinear_fused<<<grid, 256, 0, stream>>>(x, wq, scale, bias, out);
    }
}
